// SelfAttention_90477781057641
// MI455X (gfx1250) — matrix operand order repaired and checked by reading, not compiled, not run on hardware
//
#include <hip/hip_runtime.h>

typedef unsigned short u16;
typedef __attribute__((ext_vector_type(16))) __bf16 v16bf;
typedef __attribute__((ext_vector_type(8)))  float  v8f;

#define B_    2
#define S_    2048
#define D_    512
#define H_    8
#define HD_   64

union ABFrag {
  uint4 u[2];
  v16bf v;
};

__device__ __forceinline__ u16 f2bf(float f) {
  unsigned int u = __float_as_uint(f);
  u += 0x7FFFu + ((u >> 16) & 1u);     // round-to-nearest-even
  return (u16)(u >> 16);
}

__device__ __forceinline__ v8f wmma_bf16(const ABFrag& a, const ABFrag& b, v8f c) {
  return __builtin_amdgcn_wmma_f32_16x16x32_bf16(
      /*neg_a=*/false, a.v, /*neg_b=*/false, b.v,
      /*c_mod=*/(short)0, c, /*reuse_a=*/false, /*reuse_b=*/false);
}

// gfx1250 async global->LDS b128 copy: per-lane 16B, bypasses VGPRs, ASYNCcnt.
// lds_off = wave-relative LDS byte offset (low 32 bits of generic shared ptr).
__device__ __forceinline__ void async_copy_b128(const void* lds_dst,
                                                const void* gsrc) {
  unsigned int lds_off = (unsigned int)(unsigned long long)lds_dst;
  unsigned long long ga = (unsigned long long)gsrc;
  asm volatile("global_load_async_to_lds_b128 %0, %1, off"
               :: "v"(lds_off), "v"(ga)
               : "memory");
}

__device__ __forceinline__ void async_wait0() {
  asm volatile("s_wait_asynccnt 0x0" ::: "memory");
}

// ---------------------------------------------------------------------------
// Kernel 1: Q/K/V projections. grid = (M/128, N/64, 3). 256 threads (8 waves).
// Y = x @ W (f32 -> bf16), head-split [B,H,S,64]; Q pre-scaled by 0.125.
// Double-buffered LDS: next tile's f32 loads overlap this tile's WMMAs;
// conversion stores are deferred until after the WMMAs. One barrier / K-step.
// ---------------------------------------------------------------------------
__global__ __launch_bounds__(256)
void qkv_proj_kernel(const float* __restrict__ x,
                     const float* __restrict__ Wq,
                     const float* __restrict__ Wk,
                     const float* __restrict__ Wv,
                     u16* __restrict__ q_ws, u16* __restrict__ k_ws,
                     u16* __restrict__ v_ws) {
  __shared__ alignas(16) u16 As[2][128 * 40];   // 128x32, pad->40 (80B stride)
  __shared__ alignas(16) u16 Bst[2][64 * 40];   // transposed [n][k], pad->40

  const int z = blockIdx.z;
  const float* W = (z == 0) ? Wq : (z == 1) ? Wk : Wv;
  u16* dst = (z == 0) ? q_ws : (z == 1) ? k_ws : v_ws;

  const int m0 = blockIdx.x * 128;
  const int n0 = blockIdx.y * 64;
  const int tid = threadIdx.x;
  const int lane = tid & 31;
  const int w = tid >> 5;       // wave 0..7
  const int wm = w >> 1;        // 0..3
  const int wn = w & 1;         // 0..1
  const int half = lane >> 4;
  const int l16 = lane & 15;
  const int kb8 = half * 8;
  const int kb16 = half * 16;

  auto loadA = [&](int k0, float4* a) {
#pragma unroll
    for (int it = 0; it < 4; ++it) {
      int fi = tid + it * 256;
      int row = fi >> 3;
      int col4 = (fi & 7) * 4;
      a[it] = *(const float4*)(x + (size_t)(m0 + row) * D_ + k0 + col4);
    }
  };
  auto loadB = [&](int k0, float4* b) {
#pragma unroll
    for (int it = 0; it < 2; ++it) {
      int fi = tid + it * 256;
      int row = fi >> 4;
      int col4 = (fi & 15) * 4;
      b[it] = *(const float4*)(W + (size_t)(k0 + row) * D_ + n0 + col4);
    }
  };
  auto storeA = [&](int bufi, const float4* a) {
#pragma unroll
    for (int it = 0; it < 4; ++it) {
      int fi = tid + it * 256;
      int row = fi >> 3;
      int col4 = (fi & 7) * 4;
      u16* d = &As[bufi][row * 40 + col4];
      d[0] = f2bf(a[it].x); d[1] = f2bf(a[it].y);
      d[2] = f2bf(a[it].z); d[3] = f2bf(a[it].w);
    }
  };
  auto storeB = [&](int bufi, const float4* b) {
#pragma unroll
    for (int it = 0; it < 2; ++it) {
      int fi = tid + it * 256;
      int row = fi >> 4;
      int col4 = (fi & 15) * 4;
      Bst[bufi][(col4 + 0) * 40 + row] = f2bf(b[it].x);
      Bst[bufi][(col4 + 1) * 40 + row] = f2bf(b[it].y);
      Bst[bufi][(col4 + 2) * 40 + row] = f2bf(b[it].z);
      Bst[bufi][(col4 + 3) * 40 + row] = f2bf(b[it].w);
    }
  };

  v8f acc[2][2] = {};
  float4 at[4], bt[2];

  // prologue: stage K-step 0 into buffer 0
  loadA(0, at); loadB(0, bt);
  storeA(0, at); storeB(0, bt);
  __syncthreads();

  const int NK = D_ / 32;   // 16
#pragma unroll 2
  for (int ki = 0; ki < NK; ++ki) {
    const int cb = ki & 1, nb = cb ^ 1;
    if (ki + 1 < NK) { loadA((ki + 1) * 32, at); loadB((ki + 1) * 32, bt); }

    ABFrag af[2], bf[2];
#pragma unroll
    for (int mt = 0; mt < 2; ++mt) {
      int row = wm * 32 + mt * 16 + l16;
      af[mt].u[0] = *(const uint4*)&As[cb][row * 40 + kb16];
      af[mt].u[1] = *(const uint4*)&As[cb][row * 40 + kb16 + 8];
    }
#pragma unroll
    for (int nt = 0; nt < 2; ++nt) {
      int col = wn * 32 + nt * 16 + l16;
      bf[nt].u[0] = *(const uint4*)&Bst[cb][col * 40 + kb16];
      bf[nt].u[1] = *(const uint4*)&Bst[cb][col * 40 + kb16 + 8];
    }
#pragma unroll
    for (int mt = 0; mt < 2; ++mt)
#pragma unroll
      for (int nt = 0; nt < 2; ++nt)
        acc[mt][nt] = wmma_bf16(af[mt], bf[nt], acc[mt][nt]);

    if (ki + 1 < NK) { storeA(nb, at); storeB(nb, bt); }
    __syncthreads();
  }

  // epilogue -> head-split bf16 [B,H,S,64]; bake 1/sqrt(hd)=0.125 into Q
  const float scale = (z == 0) ? 0.125f : 1.0f;
#pragma unroll
  for (int mt = 0; mt < 2; ++mt)
#pragma unroll
    for (int nt = 0; nt < 2; ++nt)
#pragma unroll
      for (int r = 0; r < 8; ++r) {
        int row = m0 + wm * 32 + mt * 16 + half * 8 + r;   // b*S + s
        int col = n0 + wn * 32 + nt * 16 + l16;            // d channel
        int bb = row >> 11, s = row & (S_ - 1);
        int h = col >> 6, n = col & (HD_ - 1);
        dst[((size_t)(bb * H_ + h) * S_ + s) * HD_ + n] =
            f2bf(acc[mt][nt][r] * scale);
      }
}

// ---------------------------------------------------------------------------
// Kernel 2: flash attention. grid = (S/64, H, B). 128 threads (4 waves),
// each wave owns 16 query rows. Online softmax, key blocks of 32.
// Double-buffered K (async global->LDS) and V (VGPR-held transpose);
// one barrier per key block; async K copies overlap the WMMAs.
// ---------------------------------------------------------------------------
__global__ __launch_bounds__(128)
void flash_attn_kernel(const u16* __restrict__ qg, const u16* __restrict__ kg,
                       const u16* __restrict__ vg, u16* __restrict__ ctx_ws) {
  __shared__ alignas(16) u16 Qs[64 * 72];       // [qrow][dim], pad 64->72
  __shared__ alignas(16) u16 Ks[2][32 * 72];    // [key][dim]
  __shared__ alignas(16) u16 Vst[2][64 * 40];   // transposed [dim][key], pad
  __shared__ alignas(16) u16 Ps[4][16 * 40];    // per-wave P scratch

  const int bb = blockIdx.z, h = blockIdx.y;
  const int q0 = blockIdx.x * 64;
  const size_t base = (size_t)(bb * H_ + h) * S_ * HD_;
  const u16* Q = qg + base;
  const u16* K = kg + base;
  const u16* V = vg + base;

  const int tid = threadIdx.x;
  const int lane = tid & 31;
  const int w = tid >> 5;
  const int half = lane >> 4;
  const int l16 = lane & 15;
  const int kb8 = half * 8;
  const int kb16 = half * 16;

  auto asyncK = [&](int k0, int bufi) {
#pragma unroll
    for (int it = 0; it < 2; ++it) {
      int fi = tid + it * 128;
      int row = fi >> 3;
      int col8 = (fi & 7) * 8;
      async_copy_b128(&Ks[bufi][row * 72 + col8],
                      K + (size_t)(k0 + row) * HD_ + col8);
    }
  };
  auto loadV = [&](int k0, uint4* vt) {
#pragma unroll
    for (int it = 0; it < 2; ++it) {
      int fi = tid + it * 128;
      int row = fi >> 3;
      int col8 = (fi & 7) * 8;
      vt[it] = *(const uint4*)(V + (size_t)(k0 + row) * HD_ + col8);
    }
  };
  auto storeV = [&](int bufi, const uint4* vt) {
#pragma unroll
    for (int it = 0; it < 2; ++it) {
      int fi = tid + it * 128;
      int row = fi >> 3;             // key 0..31
      int col8 = (fi & 7) * 8;       // dim
      u16 tmp[8];
      *(uint4*)tmp = vt[it];
#pragma unroll
      for (int e = 0; e < 8; ++e) Vst[bufi][(col8 + e) * 40 + row] = tmp[e];
    }
  };

  // prologue: Q tile async (512 x b128), stage key-block 0 into buffer 0
#pragma unroll
  for (int it = 0; it < 4; ++it) {
    int fi = tid + it * 128;
    int row = fi >> 3;
    int col8 = (fi & 7) * 8;
    async_copy_b128(&Qs[row * 72 + col8],
                    Q + (size_t)(q0 + row) * HD_ + col8);
  }
  {
    uint4 vt[2];
    asyncK(0, 0);
    loadV(0, vt);
    storeV(0, vt);
  }
  async_wait0();
  __syncthreads();

  v8f ctxacc[4] = {};
  float mrun[8], lrun[8];
#pragma unroll
  for (int r = 0; r < 8; ++r) { mrun[r] = -1e30f; lrun[r] = 0.0f; }

  const int NB = S_ / 32;   // 64
#pragma unroll 2
  for (int i = 0; i < NB; ++i) {
    const int cb = i & 1, nb = cb ^ 1;
    uint4 vt[2];
    if (i + 1 < NB) {
      asyncK((i + 1) * 32, nb);        // overlaps with compute below
      loadV((i + 1) * 32, vt);         // held in VGPRs during compute
    }

    // scores: 16 query rows x 32 keys (4 WMMAs over hd=64)
    v8f sc[2] = {};
    {
      const int qrow = w * 16 + l16;
#pragma unroll
      for (int t = 0; t < 2; ++t) {
        ABFrag af, bf0, bf1;
        af.u[0] = *(const uint4*)&Qs[qrow * 72 + t * 32 + kb16];
        af.u[1] = *(const uint4*)&Qs[qrow * 72 + t * 32 + kb16 + 8];
        bf0.u[0] = *(const uint4*)&Ks[cb][l16 * 72 + t * 32 + kb16];
        bf0.u[1] = *(const uint4*)&Ks[cb][l16 * 72 + t * 32 + kb16 + 8];
        bf1.u[0] = *(const uint4*)&Ks[cb][(16 + l16) * 72 + t * 32 + kb16];
        bf1.u[1] = *(const uint4*)&Ks[cb][(16 + l16) * 72 + t * 32 + kb16 + 8];
        sc[0] = wmma_bf16(af, bf0, sc[0]);
        sc[1] = wmma_bf16(af, bf1, sc[1]);
      }
    }

    // online softmax: row stats per (vgpr r, half); reduce across 16 lanes
    float mnew[8], alpha[8];
#pragma unroll
    for (int r = 0; r < 8; ++r) {
      float mx = fmaxf(sc[0][r], sc[1][r]);
      mx = fmaxf(mx, __shfl_xor(mx, 1));
      mx = fmaxf(mx, __shfl_xor(mx, 2));
      mx = fmaxf(mx, __shfl_xor(mx, 4));
      mx = fmaxf(mx, __shfl_xor(mx, 8));
      mnew[r] = fmaxf(mrun[r], mx);
      alpha[r] = __expf(mrun[r] - mnew[r]);
      mrun[r] = mnew[r];
    }
#pragma unroll
    for (int r = 0; r < 8; ++r) {
      sc[0][r] = __expf(sc[0][r] - mnew[r]);
      sc[1][r] = __expf(sc[1][r] - mnew[r]);
      float rs = sc[0][r] + sc[1][r];
      rs += __shfl_xor(rs, 1);
      rs += __shfl_xor(rs, 2);
      rs += __shfl_xor(rs, 4);
      rs += __shfl_xor(rs, 8);
      lrun[r] = lrun[r] * alpha[r] + rs;
#pragma unroll
      for (int nt = 0; nt < 4; ++nt) ctxacc[nt][r] *= alpha[r];
    }

    // C/D-layout P -> A-layout via per-wave LDS scratch (in-order DS per wave)
#pragma unroll
    for (int nt = 0; nt < 2; ++nt)
#pragma unroll
      for (int r = 0; r < 8; ++r)
        Ps[w][(half * 8 + r) * 40 + nt * 16 + l16] = f2bf(sc[nt][r]);
    asm volatile("" ::: "memory");   // keep compiler from reordering LDS RAW

    // ctx += P(16x32) @ V(32x64)
    {
      ABFrag af;
      af.u[0] = *(const uint4*)&Ps[w][l16 * 40 + kb16];
      af.u[1] = *(const uint4*)&Ps[w][l16 * 40 + kb16 + 8];
#pragma unroll
      for (int nt = 0; nt < 4; ++nt) {
        ABFrag bf;
        int col = nt * 16 + l16;     // dim
        bf.u[0] = *(const uint4*)&Vst[cb][col * 40 + kb16];
        bf.u[1] = *(const uint4*)&Vst[cb][col * 40 + kb16 + 8];
        ctxacc[nt] = wmma_bf16(af, bf, ctxacc[nt]);
      }
    }

    if (i + 1 < NB) storeV(nb, vt);   // drain held V into next buffer
    async_wait0();                    // next K block landed in LDS
    __syncthreads();
  }

  // normalize, merge heads: ctx_ws is bf16 [B*S, D]
#pragma unroll
  for (int r = 0; r < 8; ++r) {
    float inv = 1.0f / lrun[r];
    int srow = q0 + w * 16 + half * 8 + r;
    size_t obase = ((size_t)bb * S_ + srow) * D_ + h * HD_;
#pragma unroll
    for (int nt = 0; nt < 4; ++nt)
      ctx_ws[obase + nt * 16 + l16] = f2bf(ctxacc[nt][r] * inv);
  }
}

// ---------------------------------------------------------------------------
// Kernel 3: out = ctx @ Wo. grid = (M/128, N/64). A bf16 (double-buffered
// async global->LDS), B f32->bf16 (VGPR pipeline), C f32.
// ---------------------------------------------------------------------------
__global__ __launch_bounds__(256)
void out_proj_kernel(const u16* __restrict__ ctx, const float* __restrict__ Wo,
                     float* __restrict__ out) {
  __shared__ alignas(16) u16 As[2][128 * 40];
  __shared__ alignas(16) u16 Bst[2][64 * 40];

  const int m0 = blockIdx.x * 128;
  const int n0 = blockIdx.y * 64;
  const int tid = threadIdx.x;
  const int lane = tid & 31;
  const int w = tid >> 5;
  const int wm = w >> 1;
  const int wn = w & 1;
  const int half = lane >> 4;
  const int l16 = lane & 15;
  const int kb8 = half * 8;
  const int kb16 = half * 16;

  auto asyncA = [&](int k0, int bufi) {
#pragma unroll
    for (int it = 0; it < 2; ++it) {
      int fi = tid + it * 256;
      int row = fi >> 2;
      int col8 = (fi & 3) * 8;
      async_copy_b128(&As[bufi][row * 40 + col8],
                      ctx + (size_t)(m0 + row) * D_ + k0 + col8);
    }
  };
  auto loadB = [&](int k0, float4* b) {
#pragma unroll
    for (int it = 0; it < 2; ++it) {
      int fi = tid + it * 256;
      int row = fi >> 4;
      int col4 = (fi & 15) * 4;
      b[it] = *(const float4*)(Wo + (size_t)(k0 + row) * D_ + n0 + col4);
    }
  };
  auto storeB = [&](int bufi, const float4* b) {
#pragma unroll
    for (int it = 0; it < 2; ++it) {
      int fi = tid + it * 256;
      int row = fi >> 4;
      int col4 = (fi & 15) * 4;
      Bst[bufi][(col4 + 0) * 40 + row] = f2bf(b[it].x);
      Bst[bufi][(col4 + 1) * 40 + row] = f2bf(b[it].y);
      Bst[bufi][(col4 + 2) * 40 + row] = f2bf(b[it].z);
      Bst[bufi][(col4 + 3) * 40 + row] = f2bf(b[it].w);
    }
  };

  v8f acc[2][2] = {};
  float4 bt[2];

  // prologue
  asyncA(0, 0);
  loadB(0, bt);
  storeB(0, bt);
  async_wait0();
  __syncthreads();

  const int NK = D_ / 32;
#pragma unroll 2
  for (int ki = 0; ki < NK; ++ki) {
    const int cb = ki & 1, nb = cb ^ 1;
    if (ki + 1 < NK) { asyncA((ki + 1) * 32, nb); loadB((ki + 1) * 32, bt); }

    ABFrag af[2], bf[2];
#pragma unroll
    for (int mt = 0; mt < 2; ++mt) {
      int row = wm * 32 + mt * 16 + l16;
      af[mt].u[0] = *(const uint4*)&As[cb][row * 40 + kb16];
      af[mt].u[1] = *(const uint4*)&As[cb][row * 40 + kb16 + 8];
    }
#pragma unroll
    for (int nt = 0; nt < 2; ++nt) {
      int col = wn * 32 + nt * 16 + l16;
      bf[nt].u[0] = *(const uint4*)&Bst[cb][col * 40 + kb16];
      bf[nt].u[1] = *(const uint4*)&Bst[cb][col * 40 + kb16 + 8];
    }
#pragma unroll
    for (int mt = 0; mt < 2; ++mt)
#pragma unroll
      for (int nt = 0; nt < 2; ++nt)
        acc[mt][nt] = wmma_bf16(af[mt], bf[nt], acc[mt][nt]);

    if (ki + 1 < NK) storeB(nb, bt);
    async_wait0();
    __syncthreads();
  }

#pragma unroll
  for (int mt = 0; mt < 2; ++mt)
#pragma unroll
    for (int nt = 0; nt < 2; ++nt)
#pragma unroll
      for (int r = 0; r < 8; ++r) {
        int row = m0 + wm * 32 + mt * 16 + half * 8 + r;
        int col = n0 + wn * 32 + nt * 16 + l16;
        out[(size_t)row * D_ + col] = acc[mt][nt][r];
      }
}

// ---------------------------------------------------------------------------
extern "C" void kernel_launch(void* const* d_in, const int* in_sizes, int n_in,
                              void* d_out, int out_size, void* d_ws,
                              size_t ws_size, hipStream_t stream) {
  (void)in_sizes; (void)n_in; (void)out_size; (void)ws_size;
  const float* x  = (const float*)d_in[0];
  const float* Wq = (const float*)d_in[1];
  const float* Wk = (const float*)d_in[2];
  const float* Wv = (const float*)d_in[3];
  const float* Wo = (const float*)d_in[4];
  float* out = (float*)d_out;

  const size_t n_qkv = (size_t)B_ * H_ * S_ * HD_;   // 2,097,152 elems (bf16)
  u16* q_ws = (u16*)d_ws;
  u16* k_ws = q_ws + n_qkv;
  u16* v_ws = k_ws + n_qkv;
  u16* c_ws = v_ws + n_qkv;                          // [B*S, D] bf16

  dim3 g1((B_ * S_) / 128, D_ / 64, 3);
  qkv_proj_kernel<<<g1, 256, 0, stream>>>(x, Wq, Wk, Wv, q_ws, k_ws, v_ws);

  dim3 g2(S_ / 64, H_, B_);
  flash_attn_kernel<<<g2, 128, 0, stream>>>(q_ws, k_ws, v_ws, c_ws);

  dim3 g3((B_ * S_) / 128, D_ / 64);
  out_proj_kernel<<<g3, 256, 0, stream>>>(c_ws, Wo, out);
}
